// SimpleLLM_7258494730509
// MI455X (gfx1250) — compile-verified, matrix operand order checked by reading, not run on hardware
//
#include <hip/hip_runtime.h>
#include <hip/hip_bf16.h>

// ---------------- model dims ----------------
#define S_   4096
#define V_   10000
#define E_   128
#define H_   256
#define NE_  4
#define TK_  2
#define NH_  4
#define HD_  64
#define FF_  2048
#define NL_  2
#define SCALE_ 0.125f   // 1/sqrt(64)

typedef __attribute__((ext_vector_type(16))) _Float16 v16h;
typedef __attribute__((ext_vector_type(8)))  float    v8f;

// =====================================================================
// Generic WMMA GEMM:  C[M,N] = A[M,K] * op(B) (+bias) (relu) (*scale)
//   BT=true : B is [N,K] row-major (weight, y = x @ W^T)
//   BT=false: B is [K,N] row-major (attn @ V)
// Epilogue EPI: 0 = +bias, 1 = +bias then relu, 2 = *scale (no bias)
//
// Block = 4 waves = 64(M) x 64(N) tile. Per K-step of 32:
//  - all threads stage the shared 8KB B tile into LDS with CDNA5 async
//    global->LDS copies (ASYNCcnt), then s_wait_asynccnt + barrier;
//  - each wave loads its A fragment once and issues 4 WMMAs against the
//    4 N-subtiles read from LDS (4x A reuse, 4x B reuse across waves).
// Fragment layouts per CDNA5 ISA 7.12.2 (16-bit A 16x32 / B 32x16).
// =====================================================================
template<bool BT, int EPI>
__global__ __launch_bounds__(128)
void gemm_wmma_tile(const float* __restrict__ A, int lda,
                    const float* __restrict__ B, int ldb,
                    const float* __restrict__ bias,
                    float* __restrict__ C, int ldc,
                    int N, int K, float scale)
{
    __shared__ float tb[64 * 32];          // BT: [n][k] 64x32, nonBT: [k][n] 32x64
    const int tid  = threadIdx.x;
    const int lane = threadIdx.x & 31;
    const int wave = threadIdx.x >> 5;
    const int n0   = blockIdx.x * 64;
    const int m0   = blockIdx.y * 64 + wave * 16;
    const int rA   = m0 + (lane & 15);     // A row this lane supplies
    const int kg   = (lane >> 4) << 3;     // 0 or 8 (K sub-group)
    const int Nlim = N - 1;

    v8f zero = {};
    v8f acc[4] = {zero, zero, zero, zero};

    for (int k0 = 0; k0 < K; k0 += 32) {
        // ---- async stage of B tile: 512 x 16B chunks, 4 per thread ----
        #pragma unroll
        for (int c = 0; c < 4; ++c) {
            const int chunk = tid + c * 128;              // 0..511
            const float* src;
            unsigned ldsoff;
            if (BT) {
                const int row = chunk >> 3;               // n within tile (0..63)
                const int col = (chunk & 7) * 4;          // k within tile (0..28)
                int srcRow = n0 + row; if (srcRow > Nlim) srcRow = Nlim;
                src    = B + (size_t)srcRow * ldb + (k0 + col);
                ldsoff = (unsigned)(size_t)&tb[row * 32 + col];
            } else {
                const int row = chunk >> 4;               // k within tile (0..31)
                const int col = (chunk & 15) * 4;         // n within tile (0..60)
                src    = B + (size_t)(k0 + row) * ldb + (n0 + col);
                ldsoff = (unsigned)(size_t)&tb[row * 64 + col];
            }
            asm volatile("global_load_async_to_lds_b128 %0, %1, off"
                         :: "v"(ldsoff), "v"((unsigned long long)(size_t)src)
                         : "memory");
        }
        asm volatile("s_wait_asynccnt 0" ::: "memory");
        __syncthreads();

        // ---- A fragment (f32 -> f16 on the fly) ----
        v16h a;
        const float* ap = A + (size_t)rA * lda + (k0 + kg);
        #pragma unroll
        for (int i = 0; i < 8; ++i) {
            a[i]     = (_Float16)ap[i];
            a[8 + i] = (_Float16)ap[16 + i];
        }
        if (k0 + 32 < K) __builtin_prefetch(ap + 32, 0, 1);

        // ---- 4 WMMAs against LDS-resident B ----
        #pragma unroll
        for (int nt = 0; nt < 4; ++nt) {
            const int nn = nt * 16 + (lane & 15);
            v16h b;
            if (BT) {
                #pragma unroll
                for (int i = 0; i < 8; ++i) {
                    b[i]     = (_Float16)tb[nn * 32 + kg + i];
                    b[8 + i] = (_Float16)tb[nn * 32 + kg + 16 + i];
                }
            } else {
                #pragma unroll
                for (int i = 0; i < 8; ++i) {
                    b[i]     = (_Float16)tb[(kg + i) * 64 + nn];
                    b[8 + i] = (_Float16)tb[(kg + 16 + i) * 64 + nn];
                }
            }
            acc[nt] = __builtin_amdgcn_wmma_f32_16x16x32_f16(
                false, a, false, b, (short)0, acc[nt], false, false);
        }
        __syncthreads();
    }

    // ---- epilogue (guards are wave-uniform; V tail handled at 16 cols) ----
    const int mb = m0 + ((lane >> 4) << 3);
    #pragma unroll
    for (int nt = 0; nt < 4; ++nt) {
        if (n0 + nt * 16 < N) {
            const int cB = n0 + nt * 16 + (lane & 15);
            float bv = 0.0f;
            if (EPI != 2) bv = bias[cB];
            #pragma unroll
            for (int r = 0; r < 8; ++r) {
                float v = acc[nt][r];
                if (EPI == 2) v *= scale; else v += bv;
                if (EPI == 1) v = fmaxf(v, 0.0f);
                C[(size_t)(mb + r) * ldc + cB] = v;
            }
        }
    }
}

// =====================================================================
// MoE: gate scores -> softmax -> per-token top-2 values + indices.
// (Reference quirk: expert SELECTION comes from token 0's idx.)
// =====================================================================
__global__ void moe_gate_kernel(const float* __restrict__ x, int din,
                                const float* __restrict__ gw,
                                const float* __restrict__ gb,
                                float* __restrict__ vals, int* __restrict__ idx)
{
    int s = blockIdx.x * blockDim.x + threadIdx.x;
    if (s >= S_) return;
    const float* xr = x + (size_t)s * din;
    float lg[NE_];
    #pragma unroll
    for (int e = 0; e < NE_; ++e) {
        float acc = gb[e];
        const float* w = gw + (size_t)e * din;
        for (int i = 0; i < din; ++i) acc += xr[i] * w[i];
        lg[e] = acc;
    }
    float m = fmaxf(fmaxf(lg[0], lg[1]), fmaxf(lg[2], lg[3]));
    float sum = 0.0f;
    #pragma unroll
    for (int e = 0; e < NE_; ++e) { lg[e] = __expf(lg[e] - m); sum += lg[e]; }
    float inv = 1.0f / sum;
    #pragma unroll
    for (int e = 0; e < NE_; ++e) lg[e] *= inv;
    int i0 = 0;
    #pragma unroll
    for (int e = 1; e < NE_; ++e) if (lg[e] > lg[i0]) i0 = e;
    int i1 = (i0 == 0) ? 1 : 0;
    #pragma unroll
    for (int e = 0; e < NE_; ++e) if (e != i0 && lg[e] > lg[i1]) i1 = e;
    vals[2 * s]     = lg[i0];
    vals[2 * s + 1] = lg[i1];
    idx[2 * s]      = i0;
    idx[2 * s + 1]  = i1;
}

// Dual-expert WMMA GEMM: Y[s,h] = sum_k vals[s,k]*(x @ ew[sel_k]^T + eb[sel_k])
__global__ __launch_bounds__(128)
void moe_expert_wmma(const float* __restrict__ X, int din,
                     const float* __restrict__ ew, const float* __restrict__ eb,
                     const float* __restrict__ vals, const int* __restrict__ idx,
                     float* __restrict__ Y)
{
    const int lane = threadIdx.x & 31;
    const int wave = threadIdx.x >> 5;
    const int n0   = blockIdx.x * 16;
    const int m0   = (blockIdx.y * 4 + wave) * 16;
    const int rA   = m0 + (lane & 15);
    const int cB   = n0 + (lane & 15);
    const int kg   = (lane >> 4) << 3;
    const int e0 = idx[0], e1 = idx[1];             // token 0's selection
    const float* W0 = ew + (size_t)e0 * H_ * din;
    const float* W1 = ew + (size_t)e1 * H_ * din;

    v8f acc0 = {}, acc1 = {};
    for (int k0 = 0; k0 < din; k0 += 32) {
        v16h a, b0, b1;
        const float* ap = X  + (size_t)rA * din + (k0 + kg);
        const float* p0 = W0 + (size_t)cB * din + (k0 + kg);
        const float* p1 = W1 + (size_t)cB * din + (k0 + kg);
        #pragma unroll
        for (int i = 0; i < 8; ++i) {
            a[i]      = (_Float16)ap[i];  a[8 + i]  = (_Float16)ap[16 + i];
            b0[i]     = (_Float16)p0[i];  b0[8 + i] = (_Float16)p0[16 + i];
            b1[i]     = (_Float16)p1[i];  b1[8 + i] = (_Float16)p1[16 + i];
        }
        acc0 = __builtin_amdgcn_wmma_f32_16x16x32_f16(false, a, false, b0, (short)0, acc0, false, false);
        acc1 = __builtin_amdgcn_wmma_f32_16x16x32_f16(false, a, false, b1, (short)0, acc1, false, false);
    }

    const float bb0 = eb[e0 * H_ + cB];
    const float bb1 = eb[e1 * H_ + cB];
    const int mb = m0 + ((lane >> 4) << 3);
    #pragma unroll
    for (int r = 0; r < 8; ++r) {
        int row = mb + r;
        float v0 = vals[2 * row], v1 = vals[2 * row + 1];
        Y[(size_t)row * H_ + cB] = v0 * (acc0[r] + bb0) + v1 * (acc1[r] + bb1);
    }
}

// =====================================================================
// Row softmax (attention probs), in place. One 256-thread block per row.
// =====================================================================
__global__ __launch_bounds__(256)
void softmax_rows(float* __restrict__ p0, int n)
{
    __shared__ float red[256];
    const int t = threadIdx.x;
    float* p = p0 + (size_t)blockIdx.x * n;
    float mx = -3.402823466e38f;
    for (int i = t; i < n; i += 256) mx = fmaxf(mx, p[i]);
    red[t] = mx; __syncthreads();
    for (int s2 = 128; s2 > 0; s2 >>= 1) { if (t < s2) red[t] = fmaxf(red[t], red[t + s2]); __syncthreads(); }
    float m = red[0]; __syncthreads();
    float sum = 0.0f;
    for (int i = t; i < n; i += 256) { float e = __expf(p[i] - m); p[i] = e; sum += e; }
    red[t] = sum; __syncthreads();
    for (int s2 = 128; s2 > 0; s2 >>= 1) { if (t < s2) red[t] += red[t + s2]; __syncthreads(); }
    float inv = 1.0f / red[0];
    for (int i = t; i < n; i += 256) p[i] *= inv;
}

// Residual add + LayerNorm over H=256, in place into h. One block per row.
__global__ __launch_bounds__(H_)
void add_ln_kernel(float* __restrict__ h, const float* __restrict__ y,
                   const float* __restrict__ g, const float* __restrict__ b)
{
    __shared__ float red[H_];
    const int t = threadIdx.x;
    const size_t o = (size_t)blockIdx.x * H_ + t;
    float v = h[o] + y[o];
    red[t] = v; __syncthreads();
    for (int s2 = H_ / 2; s2 > 0; s2 >>= 1) { if (t < s2) red[t] += red[t + s2]; __syncthreads(); }
    float mean = red[0] * (1.0f / H_); __syncthreads();
    float d = v - mean;
    red[t] = d * d; __syncthreads();
    for (int s2 = H_ / 2; s2 > 0; s2 >>= 1) { if (t < s2) red[t] += red[t + s2]; __syncthreads(); }
    float var = red[0] * (1.0f / H_);
    h[o] = d * rsqrtf(var + 1e-5f) * g[t] + b[t];
}

// Row log-softmax over V, in place on d_out. One block per row.
__global__ __launch_bounds__(256)
void log_softmax_rows(float* __restrict__ p0, int n)
{
    __shared__ float red[256];
    const int t = threadIdx.x;
    float* p = p0 + (size_t)blockIdx.x * n;
    float mx = -3.402823466e38f;
    for (int i = t; i < n; i += 256) mx = fmaxf(mx, p[i]);
    red[t] = mx; __syncthreads();
    for (int s2 = 128; s2 > 0; s2 >>= 1) { if (t < s2) red[t] = fmaxf(red[t], red[t + s2]); __syncthreads(); }
    float m = red[0]; __syncthreads();
    float sum = 0.0f;
    for (int i = t; i < n; i += 256) sum += __expf(p[i] - m);
    red[t] = sum; __syncthreads();
    for (int s2 = 128; s2 > 0; s2 >>= 1) { if (t < s2) red[t] += red[t + s2]; __syncthreads(); }
    float lse = m + __logf(red[0]);
    for (int i = t; i < n; i += 256) p[i] -= lse;
}

// Embedding gather: h[s,e] = emb[x[s], e]
__global__ void embed_kernel(const int* __restrict__ x,
                             const float* __restrict__ emb,
                             float* __restrict__ h)
{
    int i = blockIdx.x * blockDim.x + threadIdx.x;
    if (i >= S_ * E_) return;
    int s = i >> 7, e = i & (E_ - 1);
    h[i] = emb[(size_t)x[s] * E_ + e];
}

// =====================================================================
extern "C" void kernel_launch(void* const* d_in, const int* in_sizes, int n_in,
                              void* d_out, int out_size, void* d_ws, size_t ws_size,
                              hipStream_t stream)
{
    const int*   x         = (const int*)  d_in[0];
    const float* emb       = (const float*)d_in[1];
    const float* moe_gw[3] = {(const float*)d_in[2],  (const float*)d_in[6],  (const float*)d_in[10]};
    const float* moe_gb[3] = {(const float*)d_in[3],  (const float*)d_in[7],  (const float*)d_in[11]};
    const float* moe_ew[3] = {(const float*)d_in[4],  (const float*)d_in[8],  (const float*)d_in[12]};
    const float* moe_eb[3] = {(const float*)d_in[5],  (const float*)d_in[9],  (const float*)d_in[13]};
    const float* attn_in_w  = (const float*)d_in[14];
    const float* attn_in_b  = (const float*)d_in[15];
    const float* attn_out_w = (const float*)d_in[16];
    const float* attn_out_b = (const float*)d_in[17];
    const float* ln1_g = (const float*)d_in[18];
    const float* ln1_b = (const float*)d_in[19];
    const float* ln2_g = (const float*)d_in[20];
    const float* ln2_b = (const float*)d_in[21];
    const float* ff1_w = (const float*)d_in[22];
    const float* ff1_b = (const float*)d_in[23];
    const float* ff2_w = (const float*)d_in[24];
    const float* ff2_b = (const float*)d_in[25];
    const float* fc_w  = (const float*)d_in[26];
    const float* fc_b  = (const float*)d_in[27];
    const float* out_w = (const float*)d_in[28];
    const float* out_b = (const float*)d_in[29];
    float* out = (float*)d_out;

    // ---- workspace carve-up (floats) ----
    float* bufA   = (float*)d_ws;                 // S*H (embed uses first S*E)
    float* bufB   = bufA   + (size_t)S_ * H_;     // S*H
    float* qkv    = bufB   + (size_t)S_ * H_;     // S*3H
    float* abuf   = qkv    + (size_t)S_ * 3 * H_; // S*H
    float* ffbuf  = abuf   + (size_t)S_ * H_;     // S*FF
    float* scores = ffbuf  + (size_t)S_ * FF_;    // S*S (per-head reuse)
    float* vals   = scores + (size_t)S_ * S_;     // S*2
    int*   idx    = (int*)(vals + (size_t)S_ * 2);// S*2

    const dim3 blk(128);
    const int  GY = S_ / 64;  // block covers 64 rows of M

    // 1) embedding
    embed_kernel<<<(S_ * E_ + 255) / 256, 256, 0, stream>>>(x, emb, bufA);

    // 2) three MoE layers: bufA[E] -> bufB[H] -> bufA[H] -> bufB[H]
    const float* moe_in[3]  = {bufA, bufB, bufA};
    float*       moe_out[3] = {bufB, bufA, bufB};
    const int    moe_din[3] = {E_, H_, H_};
    for (int i = 0; i < 3; ++i) {
        moe_gate_kernel<<<(S_ + 255) / 256, 256, 0, stream>>>(
            moe_in[i], moe_din[i], moe_gw[i], moe_gb[i], vals, idx);
        moe_expert_wmma<<<dim3(H_ / 16, GY), blk, 0, stream>>>(
            moe_in[i], moe_din[i], moe_ew[i], moe_eb[i], vals, idx, moe_out[i]);
    }
    float* h = bufB;   // current hidden state [S,H]

    // 3) transformer encoder layers (post-norm, relu)
    for (int l = 0; l < NL_; ++l) {
        // qkv = h @ attn_in_w[l]^T + b     [S, 3H]
        gemm_wmma_tile<true, 0><<<dim3(3 * H_ / 64, GY), blk, 0, stream>>>(
            h, H_, attn_in_w + (size_t)l * 3 * H_ * H_, H_,
            attn_in_b + (size_t)l * 3 * H_, qkv, 3 * H_, 3 * H_, H_, 1.0f);

        for (int hh = 0; hh < NH_; ++hh) {
            const float* q = qkv + hh * HD_;
            const float* k = qkv + H_ + hh * HD_;
            const float* v = qkv + 2 * H_ + hh * HD_;
            // scores = (q @ k^T) * SCALE        [S, S]
            gemm_wmma_tile<true, 2><<<dim3(S_ / 64, GY), blk, 0, stream>>>(
                q, 3 * H_, k, 3 * H_, nullptr, scores, S_, S_, HD_, SCALE_);
            softmax_rows<<<S_, 256, 0, stream>>>(scores, S_);
            // abuf[:, hh*64:..] = scores @ v     (B row-major [S, 64])
            gemm_wmma_tile<false, 2><<<dim3(1, GY), blk, 0, stream>>>(
                scores, S_, v, 3 * H_, nullptr, abuf + hh * HD_, H_, HD_, S_, 1.0f);
        }

        // attn out projection -> bufA ; h = LN(h + bufA)
        gemm_wmma_tile<true, 0><<<dim3(H_ / 64, GY), blk, 0, stream>>>(
            abuf, H_, attn_out_w + (size_t)l * H_ * H_, H_,
            attn_out_b + (size_t)l * H_, bufA, H_, H_, H_, 1.0f);
        add_ln_kernel<<<S_, H_, 0, stream>>>(h, bufA, ln1_g + (size_t)l * H_, ln1_b + (size_t)l * H_);

        // feed-forward: ffbuf = relu(h @ ff1^T + b); bufA = ffbuf @ ff2^T + b
        gemm_wmma_tile<true, 1><<<dim3(FF_ / 64, GY), blk, 0, stream>>>(
            h, H_, ff1_w + (size_t)l * FF_ * H_, H_,
            ff1_b + (size_t)l * FF_, ffbuf, FF_, FF_, H_, 1.0f);
        gemm_wmma_tile<true, 0><<<dim3(H_ / 64, GY), blk, 0, stream>>>(
            ffbuf, FF_, ff2_w + (size_t)l * H_ * FF_, FF_,
            ff2_b + (size_t)l * H_, bufA, H_, H_, FF_, 1.0f);
        add_ln_kernel<<<S_, H_, 0, stream>>>(h, bufA, ln2_g + (size_t)l * H_, ln2_b + (size_t)l * H_);
    }

    // 4) fc -> bufA
    gemm_wmma_tile<true, 0><<<dim3(H_ / 64, GY), blk, 0, stream>>>(
        h, H_, fc_w, H_, fc_b, bufA, H_, H_, H_, 1.0f);

    // 5) logits straight into d_out, then in-place log-softmax
    gemm_wmma_tile<true, 0><<<dim3((V_ + 63) / 64, GY), blk, 0, stream>>>(
        bufA, H_, out_w, H_, out_b, out, V_, V_, H_, 1.0f);
    log_softmax_rows<<<S_, 256, 0, stream>>>(out, V_);
}
